// SemanticClusteringModule_4002909520150
// MI455X (gfx1250) — matrix operand order repaired and checked by reading, not compiled, not run on hardware
//
#include <hip/hip_runtime.h>
#include <hip/hip_bf16.h>
#include <stdint.h>

#define B_BATCH 4
#define N_PTS   4096
#define D_DIM   256
#define K_NN    16
#define ROWS    32      // rows of sim per workgroup
#define COLS    64      // columns of sim per sweep iteration
#define DCHUNK  64      // D staged per LDS B-tile chunk

typedef unsigned short u16;
typedef __attribute__((ext_vector_type(16))) __bf16 v16bf;
typedef __attribute__((ext_vector_type(8)))  float  v8f;
typedef __attribute__((ext_vector_type(4)))  unsigned int v4u;
typedef __attribute__((ext_vector_type(8)))  int v8i;
typedef __attribute__((ext_vector_type(4)))  int v4i;

union FragU { uint4 q[2]; v16bf f; };

#if defined(__has_builtin)
#if __has_builtin(__builtin_amdgcn_tensor_load_to_lds)
#define HAVE_TDM 1
#endif
#endif

// ---------------------------------------------------------------------------
// Kernel 0: one-time f32 -> (bf16 hi, bf16 lo) split, round-to-nearest-even.
// ---------------------------------------------------------------------------
__global__ void semclu_convert(const float* __restrict__ src,
                               u16* __restrict__ hi, u16* __restrict__ lo, int n) {
  int i = blockIdx.x * blockDim.x + threadIdx.x;
  if (i >= n) return;
  float x = src[i];
  unsigned xb = __float_as_uint(x);
  unsigned rh = xb + 0x7fffu + ((xb >> 16) & 1u);
  u16 h = (u16)(rh >> 16);
  float fh = __uint_as_float(((unsigned)h) << 16);
  float l = x - fh;
  unsigned lb = __float_as_uint(l);
  unsigned rl = lb + 0x7fffu + ((lb >> 16) & 1u);
  hi[i] = h;
  lo[i] = (u16)(rl >> 16);
}

// ---------------------------------------------------------------------------
// TDM: DMA a 2D tile (DCHUNK contiguous elems x COLS lines, line pitch D_DIM
// elements) of bf16 from global into LDS, compacted to [COLS][DCHUNK].
// Descriptor layout per CDNA5 ISA.
// 6-arg builtin form (amdgpu-toolchain / clang-23).
// ---------------------------------------------------------------------------
__device__ __forceinline__ void tdm_load_tile(const u16* gbase, const u16* lbase) {
#ifdef HAVE_TDM
  unsigned long long ga = (unsigned long long)(uintptr_t)gbase;   // byte addr
  unsigned int laddr = (unsigned int)(uintptr_t)lbase;            // LDS byte addr
  v4u g0 = { 1u,                                                  // count=1, user mode
             laddr,                                               // lds_addr
             (unsigned)(ga & 0xFFFFFFFFu),                        // global_addr[31:0]
             (unsigned)(((ga >> 32) & 0x1FFFFFFu) | (2u << 30)) };// addr[56:32] | type=2
  v8i g1 = { (int)(1u << 16),                 // wg_mask=0, data_size=1 (2 bytes)
             (int)((unsigned)D_DIM << 16),    // tensor_dim0 low16 (256)
             (int)((unsigned)N_PTS << 16),    // dim0 hi16=0 | tensor_dim1 low16 (4096)
             (int)((unsigned)DCHUNK << 16),   // dim1 hi16=0 | tile_dim0 (64)
             (int)COLS,                       // tile_dim1 (64) | tile_dim2=0
             (int)D_DIM,                      // tensor_dim0_stride low32 (256)
             0, 0 };                          // stride0 hi | stride1 (unused, 2D)
  v4i z4 = { 0, 0, 0, 0 };
  v8i z8 = { 0, 0, 0, 0, 0, 0, 0, 0 };
  __builtin_amdgcn_tensor_load_to_lds(g0, g1, z4, z4, z8, 0);
#else
  (void)gbase; (void)lbase;
#endif
}

// ---------------------------------------------------------------------------
// Kernel 1: fused  sim = F F^T  (bf16 hi/lo split, 3x WMMA)  +  per-row top-16.
// Grid: (N/ROWS, B), block = 256 threads = 8 waves; each wave owns a 16x16 tile.
// A fragments live in registers (128 VGPRs); B tiles stream via TDM double
// buffering; sim tile stays in LDS (never touches HBM).
// ---------------------------------------------------------------------------
__global__ __launch_bounds__(256)
void semclu_gemm_topk(const u16* __restrict__ hi, const u16* __restrict__ lo,
                      int* __restrict__ knn) {
  __shared__ u16   Bhi[2][COLS * DCHUNK];   // 2 x 8 KB
  __shared__ u16   Blo[2][COLS * DCHUNK];   // 2 x 8 KB
  __shared__ float simb[ROWS * COLS];       // 8 KB
  __shared__ float tkv[ROWS * K_NN];        // 2 KB
  __shared__ int   tki[ROWS * K_NN];        // 2 KB
  __shared__ float tkmin[ROWS];
  __shared__ int   tkmp[ROWS];

  const int tid  = threadIdx.x;
  const int lane = tid & 31;
  const int wave = tid >> 5;
  const int tr   = wave >> 2;       // 0..1 : which 16-row half
  const int tc   = wave & 3;        // 0..3 : which 16-col quarter
  const int b    = blockIdx.y;
  const int r0   = blockIdx.x * ROWS;

  const int aoff = (lane & 16) >> 1;   // A: lanes>=16 start at K+8 (staggered layout)
  const int boff = (lane & 16) >> 1;   // B: lanes>=16 start at K+8 (staggered layout)

  // ---- A fragments: whole 256-wide row slice, held in registers for the
  // entire kernel. 8 k-steps x (hi,lo) x 32B = 128 VGPRs.
  FragU a_hi[8], a_lo[8];
  {
    const size_t arow = ((size_t)b * N_PTS + r0 + tr * 16 + (lane & 15)) * D_DIM;
#pragma unroll
    for (int s = 0; s < 8; s++) {
      a_hi[s].q[0] = *(const uint4*)&hi[arow + s * 32 + aoff];
      a_hi[s].q[1] = *(const uint4*)&hi[arow + s * 32 + aoff + 16];
      a_lo[s].q[0] = *(const uint4*)&lo[arow + s * 32 + aoff];
      a_lo[s].q[1] = *(const uint4*)&lo[arow + s * 32 + aoff + 16];
    }
  }
  if (tid < ROWS) {
    for (int j = 0; j < K_NN; j++) { tkv[tid * K_NN + j] = -3.4e38f; tki[tid * K_NN + j] = 0; }
    tkmin[tid] = -3.4e38f; tkmp[tid] = 0;
  }
  __syncthreads();

  int cur = 0;
  for (int c0 = 0; c0 < N_PTS; c0 += COLS) {
    const u16* gb_hi = &hi[((size_t)b * N_PTS + c0) * D_DIM];
    const u16* gb_lo = &lo[((size_t)b * N_PTS + c0) * D_DIM];

    // Warm L2 for the next column tile (global_prefetch_b8).
    if (c0 + COLS < N_PTS)
      __builtin_prefetch(gb_hi + (size_t)COLS * D_DIM + (size_t)tid * 64, 0, 0);

#ifdef HAVE_TDM
    // Preload chunk 0 into the current buffer (TDM, one issuing wave).
    if (wave == 0) {
      tdm_load_tile(gb_hi, &Bhi[cur][0]);
      tdm_load_tile(gb_lo, &Blo[cur][0]);
    }
#endif
    v8f acc = {};
#pragma unroll
    for (int i = 0; i < 4; i++) {              // 4 D-chunks of 64
#ifdef HAVE_TDM
      if (wave == 0) {
        if (i < 3) {                           // prefetch chunk i+1 into other buffer
          tdm_load_tile(gb_hi + (size_t)(i + 1) * DCHUNK, &Bhi[cur ^ 1][0]);
          tdm_load_tile(gb_lo + (size_t)(i + 1) * DCHUNK, &Blo[cur ^ 1][0]);
          __builtin_amdgcn_s_wait_tensorcnt(2); // TDM in-order: chunk i is done
        } else {
          __builtin_amdgcn_s_wait_tensorcnt(0);
        }
      }
      __syncthreads();                         // publish chunk i to all waves
#else
      // Fallback: manual staging of chunk i into buffer `cur`.
      __syncthreads();
      for (int c = tid; c < COLS * DCHUNK / 8; c += 256) {
        int col = c >> 3; int dl = (c & 7) << 3;
        *(uint4*)&Bhi[cur][col * DCHUNK + dl] =
            *(const uint4*)&gb_hi[(size_t)col * D_DIM + i * DCHUNK + dl];
        *(uint4*)&Blo[cur][col * DCHUNK + dl] =
            *(const uint4*)&gb_lo[(size_t)col * D_DIM + i * DCHUNK + dl];
      }
      __syncthreads();
#endif
      const u16* bh = &Bhi[cur][(tc * 16 + (lane & 15)) * DCHUNK];
      const u16* bl = &Blo[cur][(tc * 16 + (lane & 15)) * DCHUNK];
#pragma unroll
      for (int ks = 0; ks < 2; ks++) {
        const int s  = i * 2 + ks;
        const int kb = ks * 32 + boff;
        FragU fbh, fbl;
        fbh.q[0] = *(const uint4*)&bh[kb];
        fbh.q[1] = *(const uint4*)&bh[kb + 16];
        fbl.q[0] = *(const uint4*)&bl[kb];
        fbl.q[1] = *(const uint4*)&bl[kb + 16];
        // sim ~= hi*hi + hi*lo + lo*hi  (fp32-class accuracy from bf16 WMMA)
        acc = __builtin_amdgcn_wmma_f32_16x16x32_bf16(false, a_hi[s].f, false, fbh.f, (short)0, acc, false, false);
        acc = __builtin_amdgcn_wmma_f32_16x16x32_bf16(false, a_hi[s].f, false, fbl.f, (short)0, acc, false, false);
        acc = __builtin_amdgcn_wmma_f32_16x16x32_bf16(false, a_lo[s].f, false, fbh.f, (short)0, acc, false, false);
      }
      __syncthreads();                         // all reads of buf `cur` done before reuse
      cur ^= 1;
    }

    // Spill this wave's 16x16 tile: lane<16 -> M=r, lane>=16 -> M=r+8.
    {
      const int crow = tr * 16 + ((lane & 16) >> 1);
      const int ccol = tc * 16 + (lane & 15);
#pragma unroll
      for (int r = 0; r < 8; r++) simb[(crow + r) * COLS + ccol] = acc[r];
    }
    __syncthreads();

    // Running top-16 (self excluded), one thread per row.
    if (tid < ROWS) {
      const int grow = r0 + tid;
      float mn = tkmin[tid]; int mp = tkmp[tid];
      float* vals = &tkv[tid * K_NN]; int* idxs = &tki[tid * K_NN];
      for (int j = 0; j < COLS; j++) {
        const int gc = c0 + j;
        if (gc == grow) continue;
        const float v = simb[tid * COLS + j];
        if (v > mn) {
          vals[mp] = v; idxs[mp] = gc;
          mn = vals[0]; mp = 0;
          for (int q = 1; q < K_NN; q++) if (vals[q] < mn) { mn = vals[q]; mp = q; }
        }
      }
      tkmin[tid] = mn; tkmp[tid] = mp;
    }
    __syncthreads();
  }

  // Sort 16 by similarity desc and emit neighbor indices.
  if (tid < ROWS) {
    const int grow = r0 + tid;
    float* vals = &tkv[tid * K_NN]; int* idxs = &tki[tid * K_NN];
    for (int a = 0; a < K_NN; a++) {
      int best = a;
      for (int c = a + 1; c < K_NN; c++) if (vals[c] > vals[best]) best = c;
      float tv = vals[a]; vals[a] = vals[best]; vals[best] = tv;
      int   ti = idxs[a]; idxs[a] = idxs[best]; idxs[best] = ti;
      knn[(((size_t)b * N_PTS) + grow) * K_NN + a] = idxs[a];
    }
  }
}

// ---------------------------------------------------------------------------
// Kernel 2: serial BFS clustering (faithful to reference; cluster == queue,
// capped at MIN_CLUSTER_SIZE=8). One block per batch, thread 0 runs the BFS.
// ---------------------------------------------------------------------------
__global__ void semclu_cluster(const int* __restrict__ knn, int* __restrict__ labels) {
  const int b = blockIdx.x;
  __shared__ unsigned char visited[N_PTS];
  for (int i = threadIdx.x; i < N_PTS; i += blockDim.x) {
    visited[i] = 0;
    labels[(size_t)b * N_PTS + i] = -1;
  }
  __syncthreads();
  if (threadIdx.x != 0) return;
  int cid = 0;
  for (int i = 0; i < N_PTS; i++) {
    if (visited[i]) continue;
    int q[8]; int qlen = 0, qh = 0;
    q[qlen++] = i; visited[i] = 1;
    bool full = false;
    while (qh < qlen && !full) {
      const int cur = q[qh++];
      const int* nbrs = knn + ((size_t)b * N_PTS + cur) * K_NN;
      for (int e = 0; e < K_NN; e++) {
        const int nb = nbrs[e];
        if (!visited[nb]) {
          visited[nb] = 1;
          q[qlen++] = nb;
          if (qlen >= 8) { full = true; break; }
        }
      }
    }
    if (qlen >= 3) {
      for (int j = 0; j < qlen; j++) labels[(size_t)b * N_PTS + q[j]] = cid;
      cid++;
    }
  }
}

// ---------------------------------------------------------------------------
extern "C" void kernel_launch(void* const* d_in, const int* in_sizes, int n_in,
                              void* d_out, int out_size, void* d_ws, size_t ws_size,
                              hipStream_t stream) {
  const float* feat = (const float*)d_in[0];
  const int total = B_BATCH * N_PTS * D_DIM;              // 4,194,304 elements

  // workspace: bf16 hi then bf16 lo (8 MB + 8 MB)
  u16* hi = (u16*)d_ws;
  u16* lo = hi + total;

  int* out    = (int*)d_out;
  int* labels = out;                                       // [B,N]
  int* knn    = out + B_BATCH * N_PTS;                     // [B,N,K]

  semclu_convert<<<(total + 255) / 256, 256, 0, stream>>>(feat, hi, lo, total);

  dim3 grid(N_PTS / ROWS, B_BATCH);                        // (128, 4)
  semclu_gemm_topk<<<grid, 256, 0, stream>>>(hi, lo, knn);

  semclu_cluster<<<B_BATCH, 256, 0, stream>>>(knn, labels);
}
